// LlamaAttention_59923383714645
// MI455X (gfx1250) — compile-verified, matrix operand order checked by reading, not run on hardware
//
#include <hip/hip_runtime.h>

// ---------------------------------------------------------------------------
// Llama attention layer for MI455X (gfx1250), bf16 WMMA path + TDM staging.
// B=1, S=2048, HID=4096, NH=32, NKV=8, HD=128.
// Cache scatter/gather is an identity for the returned tensor -> skipped.
// ---------------------------------------------------------------------------

#define S_LEN   2048
#define HID     4096
#define NH      32
#define NKV     8
#define HD      128
#define QKV_N   6144            // 4096 q + 1024 k + 1024 v
#define SCALE   0.08838834764831845f   // 1/sqrt(128)

typedef __attribute__((ext_vector_type(16))) __bf16 v16bf;
typedef __attribute__((ext_vector_type(8)))  float  v8f;
typedef __attribute__((ext_vector_type(4)))  unsigned int v4u;
typedef __attribute__((ext_vector_type(8)))  int v8i;
typedef __attribute__((ext_vector_type(4)))  int v4i;

union FragA {
    v16bf v;
    unsigned int u32[8];
    unsigned short u16[16];
};

__device__ __forceinline__ unsigned short f2bf(float f) {
    unsigned int u = __float_as_uint(f);
    unsigned int r = u + 0x7FFFu + ((u >> 16) & 1u);   // round-to-nearest-even
    return (unsigned short)(r >> 16);
}
__device__ __forceinline__ float bf2f(unsigned short h) {
    return __uint_as_float(((unsigned int)h) << 16);
}
// K-element base for VGPR j of a 16-bit 16x32 A/B fragment (wave32 layout,
// CDNA5 ISA §7.12.2): lanes 0-15 hold K 0..7 & 16..23, lanes 16-31
// hold K 8..15 & 24..31, two bf16 per VGPR.
__device__ __forceinline__ int kpat(int j, int kh) {
    int b = (j < 4) ? (j * 2) : (16 + (j - 4) * 2);
    return b + kh * 8;
}
__device__ __forceinline__ v8f wmma_bf16(v16bf a, v16bf b, v8f c) {
    return __builtin_amdgcn_wmma_f32_16x16x32_bf16(false, a, false, b,
                                                   (short)0, c, false, false);
}
__device__ __forceinline__ v8f zero8() {
    v8f z;
#pragma unroll
    for (int i = 0; i < 8; ++i) z[i] = 0.0f;
    return z;
}
// Generic pointer -> raw LDS byte offset (addrspace(3) ptrtoint).
__device__ __forceinline__ unsigned lds_off(const void* p) {
    return (unsigned)(uintptr_t)(__attribute__((address_space(3))) const void*)p;
}

// ---------------------------------------------------------------------------
// Tensor Data Mover: 2-D bf16 tile load Global -> LDS with row padding.
// D# layout per CDNA5 ISA.
// Pads LDS by 4 DWORDs after every 64 DWORDs -> 272-byte row stride,
// matching the 68-uint padded LDS layout used by the WMMA fragment loaders.
// ---------------------------------------------------------------------------
__device__ __forceinline__ void tdm_load_2d_bf16(
    unsigned lds_byte, unsigned long long gaddr,
    unsigned dim0_elems, unsigned dim1_rows, unsigned long long stride0_elems)
{
    v4u g0;
    g0[0] = 1u;                                             // count=1 (valid)
    g0[1] = lds_byte;                                       // lds_addr
    g0[2] = (unsigned)(gaddr & 0xFFFFFFFFull);              // global_addr lo
    g0[3] = (unsigned)((gaddr >> 32) & 0x01FFFFFFull)       // global_addr hi
          | (2u << 30);                                     // type=2 (image)
    v8i g1;
    g1[0] = (int)((1u << 16)        // data_size = 2 bytes
                | (1u << 20)        // pad_enable
                | (5u << 22)        // pad_interval: 64 DWORDs
                | (3u << 25));      // pad_amount: 4 DWORDs
    g1[1] = (int)((dim0_elems & 0xFFFFu) << 16);            // dim0[15:0]; bar=0
    g1[2] = (int)(((dim0_elems >> 16) & 0xFFFFu)            // dim0[31:16]
                | ((dim1_rows & 0xFFFFu) << 16));           // dim1[15:0]
    g1[3] = (int)(((dim1_rows >> 16) & 0xFFFFu)             // dim1[31:16]
                | ((dim0_elems & 0xFFFFu) << 16));          // tile_dim0
    g1[4] = (int)(dim1_rows & 0xFFFFu);                     // tile_dim1; tile2=0
    g1[5] = (int)(unsigned)(stride0_elems & 0xFFFFFFFFull); // stride0[31:0]
    g1[6] = (int)(unsigned)((stride0_elems >> 32) & 0xFFFFull); // stride0[47:32]
    g1[7] = 0;                                              // stride1 = 0
    v4i g2 = {0, 0, 0, 0};
    v4i g3 = {0, 0, 0, 0};
    v8i g4 = {0, 0, 0, 0, 0, 0, 0, 0};     // unused trailing group (6-arg form)
    __builtin_amdgcn_tensor_load_to_lds(g0, g1, g2, g3, g4, 0);
}

// ---------------------------------------------------------------------------
// Kernel 1: QKV projection.  C[2048 x 6144] = hs[2048 x 4096] * W^T
// Block tile 128x128, 8 waves (2x4), each wave 64x32 via 4x2 16x16 WMMA tiles.
// ---------------------------------------------------------------------------
__global__ __launch_bounds__(256) void qkv_gemm_kernel(
    const float* __restrict__ hs, const float* __restrict__ wq,
    const float* __restrict__ wk, const float* __restrict__ wv,
    unsigned short* __restrict__ qkv)
{
    __shared__ unsigned int ldsA[128 * 17];   // 128 rows x 16 uints + pad
    __shared__ unsigned int ldsB[128 * 17];

    const int t = threadIdx.x;
    const int lane = t & 31, w = t >> 5;
    const int wm = w >> 2, wn = w & 3;
    const int kh = lane >> 4, lr = lane & 15;
    const int m0 = blockIdx.y * 128;
    const int n0 = blockIdx.x * 128;

    const float* wptr; int nb;
    if (n0 < 4096)      { wptr = wq; nb = n0; }
    else if (n0 < 5120) { wptr = wk; nb = n0 - 4096; }
    else                { wptr = wv; nb = n0 - 5120; }

    v8f acc[8];
#pragma unroll
    for (int i = 0; i < 8; ++i) acc[i] = zero8();

    for (int kt = 0; kt < 128; ++kt) {
        const int k0 = kt * 32;
#pragma unroll
        for (int i = 0; i < 4; ++i) {
            int idx = t + i * 256;
            int row = idx >> 3, c4 = idx & 7;
            const float4 fa = *(const float4*)(hs + (size_t)(m0 + row) * HID + k0 + c4 * 4);
            ldsA[row * 17 + c4 * 2]     = (unsigned)f2bf(fa.x) | ((unsigned)f2bf(fa.y) << 16);
            ldsA[row * 17 + c4 * 2 + 1] = (unsigned)f2bf(fa.z) | ((unsigned)f2bf(fa.w) << 16);
            const float4 fb = *(const float4*)(wptr + (size_t)(nb + row) * HID + k0 + c4 * 4);
            ldsB[row * 17 + c4 * 2]     = (unsigned)f2bf(fb.x) | ((unsigned)f2bf(fb.y) << 16);
            ldsB[row * 17 + c4 * 2 + 1] = (unsigned)f2bf(fb.z) | ((unsigned)f2bf(fb.w) << 16);
            if (kt + 1 < 128) {   // hint next K tile into cache
                __builtin_prefetch(hs   + (size_t)(m0 + row) * HID + k0 + 32 + c4 * 4, 0, 1);
                __builtin_prefetch(wptr + (size_t)(nb + row) * HID + k0 + 32 + c4 * 4, 0, 1);
            }
        }
        __syncthreads();

        FragA af[4], bf[2];
#pragma unroll
        for (int mt = 0; mt < 4; ++mt) {
            int r = wm * 64 + mt * 16 + lr;
#pragma unroll
            for (int j = 0; j < 8; ++j)
                af[mt].u32[j] = ldsA[r * 17 + (kpat(j, kh) >> 1)];
        }
#pragma unroll
        for (int nt = 0; nt < 2; ++nt) {
            int r = wn * 32 + nt * 16 + lr;
#pragma unroll
            for (int j = 0; j < 8; ++j)
                bf[nt].u32[j] = ldsB[r * 17 + (kpat(j, kh) >> 1)];
        }
#pragma unroll
        for (int mt = 0; mt < 4; ++mt)
#pragma unroll
            for (int nt = 0; nt < 2; ++nt)
                acc[mt * 2 + nt] = wmma_bf16(af[mt].v, bf[nt].v, acc[mt * 2 + nt]);
        __syncthreads();
    }

#pragma unroll
    for (int mt = 0; mt < 4; ++mt)
#pragma unroll
        for (int nt = 0; nt < 2; ++nt)
#pragma unroll
            for (int j = 0; j < 8; ++j) {
                int row = m0 + wm * 64 + mt * 16 + kh * 8 + j;
                int col = n0 + wn * 32 + nt * 16 + lr;
                qkv[(size_t)row * QKV_N + col] = f2bf(acc[mt * 2 + nt][j]);
            }
}

// ---------------------------------------------------------------------------
// Kernel 2: in-place RoPE on q (cols 0..4095) and k (cols 4096..5119).
// ---------------------------------------------------------------------------
__global__ __launch_bounds__(256) void rope_kernel(
    unsigned short* __restrict__ qkv,
    const float* __restrict__ cosp, const float* __restrict__ sinp)
{
    int idx  = blockIdx.x * 256 + threadIdx.x;      // S * 40 * 64 threads
    int d    = idx & 63;
    int slot = (idx >> 6) % 40;                     // 0..31 q heads, 32..39 k heads
    int s    = idx / (40 * 64);
    int cb   = (slot < 32) ? slot * HD : 4096 + (slot - 32) * HD;
    size_t base = (size_t)s * QKV_N + cb;

    float x1 = bf2f(qkv[base + d]);
    float x2 = bf2f(qkv[base + d + 64]);
    float c1 = cosp[s * HD + d],      s1 = sinp[s * HD + d];
    float c2 = cosp[s * HD + d + 64], s2 = sinp[s * HD + d + 64];
    qkv[base + d]      = f2bf(x1 * c1 - x2 * s1);
    qkv[base + d + 64] = f2bf(x2 * c2 + x1 * s2);
}

// ---------------------------------------------------------------------------
// Kernel 3: causal flash attention.  One block = (head, 64-query tile),
// 4 waves x 16 query rows.  K/V blocks staged via TDM (tensor_load_to_lds)
// with in-DMA row padding; Q.K^T and P.V via WMMA; online softmax with
// half-wave shuffles; C->A layout shuffle of P through per-wave LDS.
// ---------------------------------------------------------------------------
__global__ __launch_bounds__(128) void attn_kernel(
    const unsigned short* __restrict__ qkv, unsigned short* __restrict__ attn)
{
    __shared__ unsigned int ldsK[64 * 68];          // 64 keys x 128 bf16, padded
    __shared__ unsigned int ldsV[64 * 68];
    __shared__ unsigned int ldsP[4 * 16 * 33];      // per-wave 16x64 bf16, padded

    const int h = blockIdx.x, qt = blockIdx.y;
    const int kvh = h >> 2;                         // GQA 4:1
    const int t = threadIdx.x, w = t >> 5, lane = t & 31;
    const int kh = lane >> 4, lr = lane & 15;
    const int q0 = qt * 64 + w * 16;
    const unsigned int* q32 = (const unsigned int*)qkv;
    unsigned short* ldsPs = (unsigned short*)ldsP;
    unsigned short* ldsVs = (unsigned short*)ldsV;
    const int wbase = w * 16 * 33;                  // uint units

    const unsigned long long gbase = (unsigned long long)(uintptr_t)qkv;
    const unsigned lkK = lds_off(&ldsK[0]);
    const unsigned lkV = lds_off(&ldsV[0]);

    // Q fragments (16 rows x 128 d), A-layout, kept in registers.
    FragA qf[4];
    {
        const int qr = q0 + lr;
#pragma unroll
        for (int kt = 0; kt < 4; ++kt)
#pragma unroll
            for (int j = 0; j < 8; ++j) {
                int k = kt * 32 + kpat(j, kh);
                qf[kt].u32[j] = q32[(size_t)qr * (QKV_N / 2) + h * 64 + (k >> 1)];
            }
    }

    v8f o[8];
#pragma unroll
    for (int i = 0; i < 8; ++i) o[i] = zero8();
    float mrow[8], lrow[8];
#pragma unroll
    for (int j = 0; j < 8; ++j) { mrow[j] = -1e30f; lrow[j] = 0.0f; }

    for (int kb = 0; kb <= qt; ++kb) {
        // Stage K and V blocks (64 x 128 bf16) via Tensor Data Mover.
        if (w == 0) {
            unsigned long long rowOff = (unsigned long long)(kb * 64) * QKV_N;
            tdm_load_2d_bf16(lkK, gbase + (rowOff + 4096 + kvh * HD) * 2ull,
                             HD, 64, QKV_N);
            tdm_load_2d_bf16(lkV, gbase + (rowOff + 5120 + kvh * HD) * 2ull,
                             HD, 64, QKV_N);
            __builtin_amdgcn_s_wait_tensorcnt(0);
        }
        __syncthreads();

        // scores = Q . K^T : 4 key sub-tiles x 4 K-steps of WMMA.
        v8f sc[4];
#pragma unroll
        for (int nt = 0; nt < 4; ++nt) sc[nt] = zero8();
#pragma unroll
        for (int nt = 0; nt < 4; ++nt)
#pragma unroll
            for (int kt = 0; kt < 4; ++kt) {
                FragA bk;
                int r = nt * 16 + lr;
#pragma unroll
                for (int j = 0; j < 8; ++j)
                    bk.u32[j] = ldsK[r * 68 + kt * 16 + (kpat(j, kh) >> 1)];
                sc[nt] = wmma_bf16(qf[kt].v, bk.v, sc[nt]);
            }

        // Online softmax (row = kh*8 + j; N values spread across 16 lanes).
        const bool diag = (kb == qt);
        float pv[4][8];
#pragma unroll
        for (int j = 0; j < 8; ++j) {
            float mx = -1e30f;
#pragma unroll
            for (int nt = 0; nt < 4; ++nt) {
                float v = sc[nt][j] * SCALE;
                if (diag) {
                    int key = kb * 64 + nt * 16 + lr;
                    int qr  = q0 + kh * 8 + j;
                    if (key > qr) v = -1e30f;
                }
                pv[nt][j] = v;
                mx = fmaxf(mx, v);
            }
#pragma unroll
            for (int off = 1; off < 16; off <<= 1)
                mx = fmaxf(mx, __shfl_xor(mx, off, 32));   // stays in half-wave
            float mn   = fmaxf(mrow[j], mx);
            float corr = __expf(mrow[j] - mn);
            mrow[j] = mn;
            float rs = 0.0f;
#pragma unroll
            for (int nt = 0; nt < 4; ++nt) {
                float e = __expf(pv[nt][j] - mn);
                pv[nt][j] = e;
                rs += e;
            }
#pragma unroll
            for (int off = 1; off < 16; off <<= 1)
                rs += __shfl_xor(rs, off, 32);
            lrow[j] = lrow[j] * corr + rs;
#pragma unroll
            for (int dt = 0; dt < 8; ++dt) o[dt][j] *= corr;
        }

        // C-layout -> A-layout shuffle of P through per-wave LDS.
#pragma unroll
        for (int j = 0; j < 8; ++j)
#pragma unroll
            for (int nt = 0; nt < 4; ++nt)
                ldsPs[wbase * 2 + (kh * 8 + j) * 66 + nt * 16 + lr] = f2bf(pv[nt][j]);
        asm volatile("s_wait_dscnt 0x0" ::: "memory");  // own-wave DS RAW fence

        // O += P . V
#pragma unroll
        for (int ks = 0; ks < 2; ++ks) {
            FragA pf;
#pragma unroll
            for (int j = 0; j < 8; ++j)
                pf.u32[j] = ldsP[wbase + lr * 33 + ks * 16 + (kpat(j, kh) >> 1)];
#pragma unroll
            for (int dt = 0; dt < 8; ++dt) {
                FragA bv;   // B[n][k] = V[key][d=n]
#pragma unroll
                for (int j = 0; j < 8; ++j) {
                    int kk = ks * 32 + kpat(j, kh);
                    bv.u16[2 * j]     = ldsVs[kk * 136 + dt * 16 + lr];
                    bv.u16[2 * j + 1] = ldsVs[(kk + 1) * 136 + dt * 16 + lr];
                }
                o[dt] = wmma_bf16(pf.v, bv.v, o[dt]);
            }
        }
        __syncthreads();
    }

    // Normalize and store (S x NH*HD) bf16.
#pragma unroll
    for (int j = 0; j < 8; ++j) {
        float inv = 1.0f / lrow[j];
        int row = q0 + kh * 8 + j;
#pragma unroll
        for (int dt = 0; dt < 8; ++dt)
            attn[(size_t)row * (NH * HD) + h * HD + dt * 16 + lr] =
                f2bf(o[dt][j] * inv);
    }
}

// ---------------------------------------------------------------------------
// Kernel 4: output projection. out[2048 x 4096] = attn[2048 x 4096] * wo^T.
// ---------------------------------------------------------------------------
__global__ __launch_bounds__(256) void out_gemm_kernel(
    const unsigned short* __restrict__ attn, const float* __restrict__ wo,
    float* __restrict__ out)
{
    __shared__ unsigned int ldsA[128 * 17];
    __shared__ unsigned int ldsB[128 * 17];

    const int t = threadIdx.x;
    const int lane = t & 31, w = t >> 5;
    const int wm = w >> 2, wn = w & 3;
    const int kh = lane >> 4, lr = lane & 15;
    const int m0 = blockIdx.y * 128;
    const int n0 = blockIdx.x * 128;
    const unsigned int* a32 = (const unsigned int*)attn;

    v8f acc[8];
#pragma unroll
    for (int i = 0; i < 8; ++i) acc[i] = zero8();

    for (int kt = 0; kt < 128; ++kt) {
        const int k0 = kt * 32;
#pragma unroll
        for (int i = 0; i < 8; ++i) {       // A: bf16 pass-through
            int idx = t + i * 256;
            int row = idx >> 4, cu = idx & 15;
            ldsA[row * 17 + cu] = a32[(size_t)(m0 + row) * (HID / 2) + kt * 16 + cu];
        }
#pragma unroll
        for (int i = 0; i < 4; ++i) {       // B: f32 -> bf16
            int idx = t + i * 256;
            int row = idx >> 3, c4 = idx & 7;
            const float4 fb = *(const float4*)(wo + (size_t)(n0 + row) * HID + k0 + c4 * 4);
            ldsB[row * 17 + c4 * 2]     = (unsigned)f2bf(fb.x) | ((unsigned)f2bf(fb.y) << 16);
            ldsB[row * 17 + c4 * 2 + 1] = (unsigned)f2bf(fb.z) | ((unsigned)f2bf(fb.w) << 16);
            if (kt + 1 < 128)
                __builtin_prefetch(wo + (size_t)(n0 + row) * HID + k0 + 32 + c4 * 4, 0, 1);
        }
        __syncthreads();

        FragA af[4], bf[2];
#pragma unroll
        for (int mt = 0; mt < 4; ++mt) {
            int r = wm * 64 + mt * 16 + lr;
#pragma unroll
            for (int j = 0; j < 8; ++j)
                af[mt].u32[j] = ldsA[r * 17 + (kpat(j, kh) >> 1)];
        }
#pragma unroll
        for (int nt = 0; nt < 2; ++nt) {
            int r = wn * 32 + nt * 16 + lr;
#pragma unroll
            for (int j = 0; j < 8; ++j)
                bf[nt].u32[j] = ldsB[r * 17 + (kpat(j, kh) >> 1)];
        }
#pragma unroll
        for (int mt = 0; mt < 4; ++mt)
#pragma unroll
            for (int nt = 0; nt < 2; ++nt)
                acc[mt * 2 + nt] = wmma_bf16(af[mt].v, bf[nt].v, acc[mt * 2 + nt]);
        __syncthreads();
    }

#pragma unroll
    for (int mt = 0; mt < 4; ++mt)
#pragma unroll
        for (int nt = 0; nt < 2; ++nt)
#pragma unroll
            for (int j = 0; j < 8; ++j) {
                int row = m0 + wm * 64 + mt * 16 + kh * 8 + j;
                int col = n0 + wn * 32 + nt * 16 + lr;
                out[(size_t)row * HID + col] = acc[mt * 2 + nt][j];
            }
}

// ---------------------------------------------------------------------------
extern "C" void kernel_launch(void* const* d_in, const int* in_sizes, int n_in,
                              void* d_out, int out_size, void* d_ws, size_t ws_size,
                              hipStream_t stream)
{
    (void)in_sizes; (void)n_in; (void)out_size; (void)ws_size;
    const float* hs   = (const float*)d_in[0];
    const float* cosp = (const float*)d_in[1];
    const float* sinp = (const float*)d_in[2];
    const float* wq   = (const float*)d_in[3];
    const float* wk   = (const float*)d_in[4];
    const float* wv   = (const float*)d_in[5];
    const float* wo   = (const float*)d_in[6];
    // d_in[7..9]: k_cache, v_cache, block_ids — identity for the returned
    // tensor (block_ids is a permutation; caches are not outputs).

    unsigned short* qkv  = (unsigned short*)d_ws;                      // 24 MB bf16
    unsigned short* attn = qkv + (size_t)S_LEN * QKV_N;                // 16 MB bf16

    qkv_gemm_kernel<<<dim3(QKV_N / 128, S_LEN / 128), 256, 0, stream>>>(
        hs, wq, wk, wv, qkv);
    rope_kernel<<<(S_LEN * (NH + NKV) * (HD / 2)) / 256, 256, 0, stream>>>(
        qkv, cosp, sinp);
    attn_kernel<<<dim3(NH, S_LEN / 64), 128, 0, stream>>>(qkv, attn);
    out_gemm_kernel<<<dim3(HID / 128, S_LEN / 128), 256, 0, stream>>>(
        attn, wo, (float*)d_out);
}
